// MultiScaleAttention_17403207483402
// MI455X (gfx1250) — compile-verified, matrix operand order checked by reading, not run on hardware
//
#include <hip/hip_runtime.h>
#include <stdint.h>

// ---------------------------------------------------------------------------
// MI455X (gfx1250, wave32) multi-scale attention pipeline.
// Heavy contractions: v_wmma_f32_16x16x32_bf16 (f32 accumulate).
// V-tile staging: global_load_async_to_lds_b128 (ASYNCcnt) overlapping WMMA.
// ---------------------------------------------------------------------------

typedef __attribute__((ext_vector_type(16))) __bf16        v16bf;
typedef __attribute__((ext_vector_type(8)))  float         v8f;
typedef __attribute__((ext_vector_type(4)))  unsigned int  v4u;
typedef unsigned short u16;

#define NTOK  6400     // H*W = 80*80
#define CDIM  256
#define DHEAD 32

__device__ __forceinline__ u16 f2bf(float x) {
    unsigned int u = __float_as_uint(x);
    u = u + 0x7FFFu + ((u >> 16) & 1u);   // round-to-nearest-even
    return (u16)(u >> 16);
}

// A/B fragment load for 16x16x32 bf16 WMMA.
// lane%16 = row (A) / col (B); K halves: [8g..8g+7] and [16+8g..16+8g+7].
__device__ __forceinline__ v16bf load_frag(const u16* p, int lane) {
    int g = (lane >> 4) & 1;
    union { v16bf v; v4u u[2]; } f;
    f.u[0] = *(const v4u*)(p + 8 * g);
    f.u[1] = *(const v4u*)(p + 16 + 8 * g);
    return f.v;
}

// B fragment gather for NN product: base = 32x32 row-major u16 tile in LDS,
// lane's column = col; element i -> K = 16*(i/8) + 8*(lane/16) + i%8.
__device__ __forceinline__ v16bf load_col_frag(const u16* base, int col, int lane) {
    int g = (lane >> 4) & 1;
    union { v16bf v; u16 e[16]; } f;
#pragma unroll
    for (int i = 0; i < 16; ++i) {
        int K = ((i >> 3) << 4) + 8 * g + (i & 7);
        f.e[i] = base[K * 32 + col];
    }
    return f.v;
}

__device__ __forceinline__ v8f wmma_bf(v16bf a, v16bf b, v8f c) {
    return __builtin_amdgcn_wmma_f32_16x16x32_bf16(false, a, false, b,
                                                   (short)0, c, false, false);
}

// Per-lane async 16B global -> LDS copy (CDNA5, tracked with ASYNCcnt).
__device__ __forceinline__ void async_copy16(const void* gsrc, void* lds_dst) {
    unsigned lds_off = (unsigned)(uintptr_t)lds_dst;           // addr[31:0] = LDS byte addr
    unsigned long long ga = (unsigned long long)(uintptr_t)gsrc;
    asm volatile("global_load_async_to_lds_b128 %0, %1, off"
                 :: "v"(lds_off), "v"(ga) : "memory");
}
__device__ __forceinline__ void wait_async0() {
    asm volatile("s_wait_asynccnt 0x0" ::: "memory");
}

// ---------------------------------------------------------------------------
// NT GEMM:  C[M,Nn] = A[M,K] (bf16) x W[Nn,K] (bf16)^T  + bias / relu.
// Block = 8 waves -> 128x64 tile; wave = 16x64.  K%32==0, Nn%64==0, M%16==0.
// ---------------------------------------------------------------------------
__global__ void gemm_nt_wmma(const u16* __restrict__ A, const u16* __restrict__ Bw,
                             const float* __restrict__ bias,
                             float* __restrict__ Cf, u16* __restrict__ Cb,
                             int M, int Nn, int K, int relu) {
    int lane = threadIdx.x & 31;
    int wave = threadIdx.x >> 5;
    int m0 = (blockIdx.x * 8 + wave) * 16;
    if (m0 >= M) return;
    int n0 = blockIdx.y * 64;
    int c = lane & 15, h = lane >> 4;

    const u16* arow = A + (size_t)(m0 + c) * K;
    const u16* brow0 = Bw + (size_t)(n0 + c) * K;
    v8f acc0 = {}, acc1 = {}, acc2 = {}, acc3 = {};
    for (int k0 = 0; k0 < K; k0 += 32) {
        // speculative prefetch of the next K slice (dropped silently if OOB)
        __builtin_prefetch(arow + k0 + 32, 0, 1);
        __builtin_prefetch(brow0 + k0 + 32, 0, 1);
        v16bf af = load_frag(arow + k0, lane);
        v16bf b0 = load_frag(Bw + (size_t)(n0 +  0 + c) * K + k0, lane);
        v16bf b1 = load_frag(Bw + (size_t)(n0 + 16 + c) * K + k0, lane);
        v16bf b2 = load_frag(Bw + (size_t)(n0 + 32 + c) * K + k0, lane);
        v16bf b3 = load_frag(Bw + (size_t)(n0 + 48 + c) * K + k0, lane);
        acc0 = wmma_bf(af, b0, acc0);
        acc1 = wmma_bf(af, b1, acc1);
        acc2 = wmma_bf(af, b2, acc2);
        acc3 = wmma_bf(af, b3, acc3);
    }
#pragma unroll
    for (int t = 0; t < 4; ++t) {
        v8f acc = (t == 0) ? acc0 : (t == 1) ? acc1 : (t == 2) ? acc2 : acc3;
        int col = n0 + t * 16 + c;
        float bv = bias ? bias[col] : 0.f;
#pragma unroll
        for (int r = 0; r < 8; ++r) {
            float v = acc[r] + bv;
            if (relu) v = fmaxf(v, 0.f);
            size_t idx = (size_t)(m0 + r + 8 * h) * Nn + col;
            if (Cf) Cf[idx] = v;
            if (Cb) Cb[idx] = f2bf(v);
        }
    }
}

// ---------------------------------------------------------------------------
// Windowed / global flash attention (p==0: global over kv_sr rows).
// One wave owns 16 query rows; keys processed in chunks of 32 with streaming
// softmax.  V chunk is staged into LDS with async b128 copies that overlap
// the QK^T WMMAs; probabilities go through LDS as bf16 A-fragments.
// ---------------------------------------------------------------------------
__device__ __forceinline__ int patch_tok(int s, int p) {
    if (p == 0) return s;
    int pp = p * p;
    int win = s / pp, t = s - win * pp;
    int wpr = 80 / p;
    int wy = win / wpr, wx = win - wy * wpr;
    int iy = t / p,     ix = t - iy * p;
    return (wy * p + iy) * 80 + (wx * p + ix);
}

__global__ void attn_win_wmma(const u16* __restrict__ qb, const u16* __restrict__ kb,
                              const u16* __restrict__ vb, int kstride,
                              u16* __restrict__ ob, int p, int Lk, float scale) {
    __shared__ float Sbuf[8][16][33];
    __shared__ __align__(16) u16 Pbuf[8][16][32];
    __shared__ __align__(16) u16 Vrows[8][32][32];   // [k-in-chunk][dim], row-major
    __shared__ float rowM[8][16], rowL[8][16], rowA[8][16];

    int lane = threadIdx.x & 31;
    int wave = threadIdx.x >> 5;
    int head = blockIdx.y;
    int s0 = (blockIdx.x * 8 + wave) * 16;
    int pp = p ? p * p : 0;
    int win = p ? s0 / pp : 0;
    int c = lane & 15, h = lane >> 4;

    int qtok = patch_tok(s0 + c, p);
    v16bf qa = load_frag(qb + (size_t)qtok * CDIM + head * DHEAD, lane);

    if (lane < 16) { rowM[wave][lane] = -1e30f; rowL[wave][lane] = 0.f; }
    v8f acc0 = {}, acc1 = {};

    for (int k0 = 0; k0 < Lk; k0 += 32) {
        int valid = Lk - k0; if (valid > 32) valid = 32;
        __syncthreads();   // fence: previous-iteration LDS reads complete
        // ---- async-stage V chunk: 32 rows x 64B = 128 x 16B chunks / wave ----
#pragma unroll
        for (int j = 0; j < 4; ++j) {
            int ch = lane + 32 * j;
            int row = ch >> 2, part = ch & 3;
            int kk = (row < valid) ? row : (valid - 1);
            int kt = p ? patch_tok(win * pp + k0 + kk, p) : (k0 + kk);
            const u16* gsrc = vb + (size_t)kt * kstride + head * DHEAD + part * 8;
            async_copy16(gsrc, &Vrows[wave][row][part * 8]);
        }
        // ---- scores: S = scale * Q K^T (overlaps with the async copies) ----
        {
            int kk0 = c;      if (kk0 >= valid) kk0 = valid - 1;
            int kk1 = 16 + c; if (kk1 >= valid) kk1 = valid - 1;
            int kt0 = p ? patch_tok(win * pp + k0 + kk0, p) : (k0 + kk0);
            int kt1 = p ? patch_tok(win * pp + k0 + kk1, p) : (k0 + kk1);
            v16bf kf0 = load_frag(kb + (size_t)kt0 * kstride + head * DHEAD, lane);
            v16bf kf1 = load_frag(kb + (size_t)kt1 * kstride + head * DHEAD, lane);
            v8f z = {};
            v8f c0 = wmma_bf(qa, kf0, z);
            v8f c1 = wmma_bf(qa, kf1, z);
#pragma unroll
            for (int r = 0; r < 8; ++r) {
                Sbuf[wave][r + 8 * h][c]      = c0[r] * scale;
                Sbuf[wave][r + 8 * h][16 + c] = c1[r] * scale;
            }
        }
        __syncthreads();
        // ---- streaming softmax (one lane per query row) ----
        if (lane < 16) {
            int row = lane;
            float mold = rowM[wave][row], mnew = mold;
            for (int j = 0; j < valid; ++j) mnew = fmaxf(mnew, Sbuf[wave][row][j]);
            float alpha = __expf(mold - mnew);
            float sum = 0.f;
            for (int j = 0; j < 32; ++j) {
                float pv = (j < valid) ? __expf(Sbuf[wave][row][j] - mnew) : 0.f;
                sum += pv;
                Pbuf[wave][row][j] = f2bf(pv);
            }
            rowM[wave][row] = mnew;
            rowL[wave][row] = rowL[wave][row] * alpha + sum;
            rowA[wave][row] = alpha;
        }
        __syncthreads();
        wait_async0();    // wave's own V staging must be in LDS before reading
        // ---- P x V (two 16x16 output tiles over the 32 head dims) ----
        v16bf pf  = load_frag(&Pbuf[wave][c][0], lane);
        v16bf vf0 = load_col_frag(&Vrows[wave][0][0], c, lane);
        v16bf vf1 = load_col_frag(&Vrows[wave][0][0], 16 + c, lane);
        v8f z = {};
        v8f p0 = wmma_bf(pf, vf0, z);
        v8f p1 = wmma_bf(pf, vf1, z);
#pragma unroll
        for (int r = 0; r < 8; ++r) {
            float al = rowA[wave][r + 8 * h];
            acc0[r] = acc0[r] * al + p0[r];
            acc1[r] = acc1[r] * al + p1[r];
        }
    }
    // ---- normalize and scatter back to token order ----
#pragma unroll
    for (int r = 0; r < 8; ++r) {
        int s = s0 + r + 8 * h;
        int tok = patch_tok(s, p);
        float invl = 1.f / rowL[wave][r + 8 * h];
        ob[(size_t)tok * CDIM + head * DHEAD + c]      = f2bf(acc0[r] * invl);
        ob[(size_t)tok * CDIM + head * DHEAD + 16 + c] = f2bf(acc1[r] * invl);
    }
}

// ---------------------------------------------------------------------------
// Glue kernels
// ---------------------------------------------------------------------------
__global__ void cvt_bf(const float* __restrict__ s, u16* __restrict__ d, int n) {
    int i = blockIdx.x * 256 + threadIdx.x;
    if (i < n) d[i] = f2bf(s[i]);
}

__global__ void zero_u16k(u16* d, int n) {
    int i = blockIdx.x * 256 + threadIdx.x;
    if (i < n) d[i] = 0;
}

// im2col for the 4x4/stride-4 reduction conv: Asr[m][c*16+ky*4+kx]
__global__ void gather_sr_k(const u16* __restrict__ xbf, u16* __restrict__ Asr) {
    int i = blockIdx.x * 256 + threadIdx.x;
    if (i >= 400 * 4096) return;
    int m = i >> 12, r = i & 4095;
    int cc = r >> 4, t = r & 15, ky = t >> 2, kx = t & 3;
    int Y = m / 20, X = m - Y * 20;
    int tok = (Y * 4 + ky) * 80 + (X * 4 + kx);
    Asr[i] = xbf[(size_t)tok * CDIM + cc];
}

__global__ void layernorm_bf_k(const float* __restrict__ X, const float* __restrict__ g,
                               const float* __restrict__ b, u16* __restrict__ Y) {
    __shared__ float s1[256], s2[256];
    int row = blockIdx.x, cc = threadIdx.x;
    float v = X[(size_t)row * 256 + cc];
    s1[cc] = v; s2[cc] = v * v;
    __syncthreads();
    for (int off = 128; off > 0; off >>= 1) {
        if (cc < off) { s1[cc] += s1[cc + off]; s2[cc] += s2[cc + off]; }
        __syncthreads();
    }
    float mean = s1[0] * (1.f / 256.f);
    float var  = s2[0] * (1.f / 256.f) - mean * mean;
    float y = (v - mean) * rsqrtf(var + 1e-5f) * g[cc] + b[cc];
    Y[(size_t)row * 256 + cc] = f2bf(y);
}

__global__ void ew_add_k(const float* a, const float* b2, float* of, u16* ob, int n) {
    int i = blockIdx.x * 256 + threadIdx.x;
    if (i < n) { float v = a[i] + b2[i]; of[i] = v; ob[i] = f2bf(v); }
}

// pooled branch of _ms: mean over tokens -> 1x1 conv -> relu -> 1x1 conv
__global__ void ms_global_k(const float* __restrict__ xa,
                            const float* __restrict__ w1, const float* __restrict__ b1,
                            const float* __restrict__ w2, const float* __restrict__ b2,
                            float* __restrict__ gout) {
    __shared__ float mean[256];
    __shared__ float t1[64];
    int cc = threadIdx.x;
    float s = 0.f;
    for (int n = 0; n < NTOK; ++n) s += xa[(size_t)n * 256 + cc];
    mean[cc] = s * (1.f / (float)NTOK);
    __syncthreads();
    if (cc < 64) {
        float a = b1[cc];
        for (int k = 0; k < 256; ++k) a += mean[k] * w1[cc * 256 + k];
        t1[cc] = fmaxf(a, 0.f);
    }
    __syncthreads();
    float o = b2[cc];
    for (int k = 0; k < 64; ++k) o += t1[k] * w2[cc * 64 + k];
    gout[cc] = o;
}

__global__ void ew_combine_k(const float* __restrict__ x, const float* __restrict__ res,
                             const float* __restrict__ msl, const float* __restrict__ gvec,
                             float* __restrict__ of, u16* __restrict__ ob, int n) {
    int i = blockIdx.x * 256 + threadIdx.x;
    if (i >= n) return;
    float wl = msl[i] + gvec[i & 255];
    float w = 1.f / (1.f + __expf(-wl));
    float v = x[i] * w + res[i] * (1.f - w);
    of[i] = v;
    if (ob) ob[i] = f2bf(v);
}

// ---------------------------------------------------------------------------
// Host orchestration
// ---------------------------------------------------------------------------
extern "C" void kernel_launch(void* const* d_in, const int* in_sizes, int n_in,
                              void* d_out, int out_size, void* d_ws, size_t ws_size,
                              hipStream_t stream) {
    (void)in_sizes; (void)n_in; (void)out_size; (void)ws_size;
    const float* x      = (const float*)d_in[0];
    const float* Wq     = (const float*)d_in[1];
    const float* Wkv    = (const float*)d_in[2];
    const float* Wsr    = (const float*)d_in[3];
    const float* bsr    = (const float*)d_in[4];
    const float* ln_g   = (const float*)d_in[5];
    const float* ln_b   = (const float*)d_in[6];
    const float* Wproj  = (const float*)d_in[7];
    const float* bproj  = (const float*)d_in[8];
    const float* Wfinal = (const float*)d_in[9];
    const float* bfinal = (const float*)d_in[10];
    const float* la1_w1 = (const float*)d_in[11];
    const float* la1_b1 = (const float*)d_in[12];
    const float* la1_w2 = (const float*)d_in[13];
    const float* la1_b2 = (const float*)d_in[14];
    const float* ga1_w1 = (const float*)d_in[15];
    const float* ga1_b1 = (const float*)d_in[16];
    const float* ga1_w2 = (const float*)d_in[17];
    const float* ga1_b2 = (const float*)d_in[18];
    const float* la2_w1 = (const float*)d_in[19];
    const float* la2_b1 = (const float*)d_in[20];
    const float* la2_w2 = (const float*)d_in[21];
    const float* la2_b2 = (const float*)d_in[22];
    const float* ga2_w1 = (const float*)d_in[23];
    const float* ga2_b1 = (const float*)d_in[24];
    const float* ga2_w2 = (const float*)d_in[25];
    const float* ga2_b2 = (const float*)d_in[26];

    const int NC = NTOK * CDIM;          // 1,638,400

    char* wsp = (char*)d_ws;
    auto alloc = [&](size_t bytes) -> char* {
        char* pc = wsp;
        wsp += (bytes + 255) & ~(size_t)255;
        return pc;
    };
    u16*   xbf     = (u16*)alloc((size_t)NC * 2);
    u16*   WqB     = (u16*)alloc((size_t)65536 * 2);
    u16*   WkvB    = (u16*)alloc((size_t)131072 * 2);
    u16*   WsrB    = (u16*)alloc((size_t)1048576 * 2);
    u16*   WprojB  = (u16*)alloc((size_t)65536 * 2);
    u16*   WfinalB = (u16*)alloc((size_t)65536 * 2);
    u16*   la1w1B  = (u16*)alloc((size_t)16384 * 2);
    u16*   la1w2B  = (u16*)alloc((size_t)16384 * 2);
    u16*   la2w1B  = (u16*)alloc((size_t)16384 * 2);
    u16*   la2w2B  = (u16*)alloc((size_t)16384 * 2);
    u16*   qbf     = (u16*)alloc((size_t)NC * 2);
    u16*   kvbf    = (u16*)alloc((size_t)NC * 2 * 2);
    u16*   AsrB    = (u16*)alloc((size_t)400 * 4096 * 2);
    float* sr_f    = (float*)alloc((size_t)400 * 256 * 4);
    u16*   xsrB    = (u16*)alloc((size_t)400 * 256 * 2);
    u16*   kvsrB   = (u16*)alloc((size_t)416 * 512 * 2);
    u16*   attnbf  = (u16*)alloc((size_t)NC * 2);
    float* A0      = (float*)alloc((size_t)NC * 4);
    float* A1      = (float*)alloc((size_t)NC * 4);
    float* A2      = (float*)alloc((size_t)NC * 4);
    float* A3      = (float*)alloc((size_t)NC * 4);
    float* T1      = (float*)alloc((size_t)NC * 4);
    float* T2      = (float*)alloc((size_t)NC * 4);
    float* xa_f    = (float*)alloc((size_t)NC * 4);
    u16*   xa_bf   = (u16*)alloc((size_t)NC * 2);
    float* xi_f    = (float*)alloc((size_t)NC * 4);
    u16*   xi_bf   = (u16*)alloc((size_t)NC * 2);
    u16*   ms1_bf  = (u16*)alloc((size_t)NTOK * 64 * 2);
    float* msl_f   = (float*)alloc((size_t)NC * 4);
    float* gvec    = (float*)alloc((size_t)256 * 4);
    u16*   outbf   = (u16*)alloc((size_t)NC * 2);

    auto cvt = [&](const float* s, u16* d, int n) {
        cvt_bf<<<dim3((n + 255) / 256), dim3(256), 0, stream>>>(s, d, n);
    };

    // 1) bf16 conversions of activations and all GEMM weights
    cvt(x, xbf, NC);
    cvt(Wq, WqB, 65536);
    cvt(Wkv, WkvB, 131072);
    cvt(Wsr, WsrB, 1048576);
    cvt(Wproj, WprojB, 65536);
    cvt(Wfinal, WfinalB, 65536);
    cvt(la1_w1, la1w1B, 16384);
    cvt(la1_w2, la1w2B, 16384);
    cvt(la2_w1, la2w1B, 16384);
    cvt(la2_w2, la2w2B, 16384);

    // 2) q and full-resolution kv projections (bf16 outputs for attention)
    gemm_nt_wmma<<<dim3(50, 4), dim3(256), 0, stream>>>(xbf, WqB, nullptr,
        nullptr, qbf, NTOK, 256, 256, 0);
    gemm_nt_wmma<<<dim3(50, 8), dim3(256), 0, stream>>>(xbf, WkvB, nullptr,
        nullptr, kvbf, NTOK, 512, 256, 0);

    // 3) spatial reduction: im2col gather + GEMM + LayerNorm + kv_sr projection
    gather_sr_k<<<dim3((400 * 4096 + 255) / 256), dim3(256), 0, stream>>>(xbf, AsrB);
    gemm_nt_wmma<<<dim3(4, 4), dim3(256), 0, stream>>>(AsrB, WsrB, bsr,
        sr_f, nullptr, 400, 256, 4096, 0);
    layernorm_bf_k<<<dim3(400), dim3(256), 0, stream>>>(sr_f, ln_g, ln_b, xsrB);
    zero_u16k<<<dim3((416 * 512 + 255) / 256), dim3(256), 0, stream>>>(kvsrB, 416 * 512);
    gemm_nt_wmma<<<dim3(4, 8), dim3(256), 0, stream>>>(xsrB, WkvB, nullptr,
        nullptr, kvsrB, 400, 512, 256, 0);

    // 4) attentions + projections
    const float scale = 0.17677669529663687f;   // 1/sqrt(32)
    attn_win_wmma<<<dim3(50, 8), dim3(256), 0, stream>>>(qbf, kvsrB, kvsrB + 256,
        512, attnbf, 0, 400, scale);
    gemm_nt_wmma<<<dim3(50, 4), dim3(256), 0, stream>>>(attnbf, WprojB, bproj,
        A0, nullptr, NTOK, 256, 256, 0);

    const int ps[3] = {4, 8, 40};
    float* Ap[3] = {A1, A2, A3};
    for (int ii = 0; ii < 3; ++ii) {
        attn_win_wmma<<<dim3(50, 8), dim3(256), 0, stream>>>(qbf, kvbf, kvbf + 256,
            512, attnbf, ps[ii], ps[ii] * ps[ii], scale);
        gemm_nt_wmma<<<dim3(50, 4), dim3(256), 0, stream>>>(attnbf, WprojB, bproj,
            Ap[ii], nullptr, NTOK, 256, 256, 0);
    }

    // 5) iAFF fusion chain
    auto run_ms = [&](const u16* inbf, const u16* w1b, const float* b1,
                      const u16* w2b, const float* b2) {
        gemm_nt_wmma<<<dim3(50, 1), dim3(256), 0, stream>>>(inbf, w1b, b1,
            nullptr, ms1_bf, NTOK, 64, 256, 1);
        gemm_nt_wmma<<<dim3(50, 4), dim3(256), 0, stream>>>(ms1_bf, w2b, b2,
            msl_f, nullptr, NTOK, 256, 64, 0);
    };
    auto run_iaff = [&](const float* xin, const float* res, float* outp, u16* outb) {
        ew_add_k<<<dim3(6400), dim3(256), 0, stream>>>(xin, res, xa_f, xa_bf, NC);
        run_ms(xa_bf, la1w1B, la1_b1, la1w2B, la1_b2);
        ms_global_k<<<dim3(1), dim3(256), 0, stream>>>(xa_f, ga1_w1, ga1_b1,
            ga1_w2, ga1_b2, gvec);
        ew_combine_k<<<dim3(6400), dim3(256), 0, stream>>>(xin, res, msl_f, gvec,
            xi_f, xi_bf, NC);
        run_ms(xi_bf, la2w1B, la2_b1, la2w2B, la2_b2);
        ms_global_k<<<dim3(1), dim3(256), 0, stream>>>(xi_f, ga2_w1, ga2_b1,
            ga2_w2, ga2_b2, gvec);
        ew_combine_k<<<dim3(6400), dim3(256), 0, stream>>>(xin, res, msl_f, gvec,
            outp, outb, NC);
    };
    run_iaff(A1, A2, T1, nullptr);
    run_iaff(T1, A3, T2, nullptr);
    run_iaff(T2, A0, T1, outbf);     // last combine also emits bf16 for final GEMM

    // 6) final projection -> d_out (f32)
    gemm_nt_wmma<<<dim3(50, 4), dim3(256), 0, stream>>>(outbf, WfinalB, bfinal,
        (float*)d_out, nullptr, NTOK, 256, 256, 0);
}
